// MemoryUnit_23905787969671
// MI455X (gfx1250) — matrix operand order repaired and checked by reading, not compiled, not run on hardware
//
#include <hip/hip_runtime.h>
#include <hip/hip_bf16.h>

// ---------------------------------------------------------------------------
// MemoryUnit scan for MI455X (gfx1250).
// One workgroup per batch element (64 blocks x 256 threads = 8 wave32s).
// Recurrent state (M, pers) lives in LDS; 16xD GEMMs use
// v_wmma_f32_16x16x32_bf16 (f32 accumulate); weights pre-swizzled into WMMA
// B-fragment-major bf16 in workspace (contiguous 32B per lane). Skinny GEMVs
// use transposed f32 weight copies so each thread streams contiguous b128
// loads. Per-step-invariant scalars are hoisted out of the recurrence loop.
// ---------------------------------------------------------------------------

typedef __attribute__((ext_vector_type(16))) __bf16 v16bf;
typedef __attribute__((ext_vector_type(8)))  __bf16 v8bf;
typedef __attribute__((ext_vector_type(8)))  float  v8f;

constexpr int Bsz = 64;   // batch
constexpr int D   = 256;  // model dim
constexpr int K   = 16;   // slots

union V16U { v16bf v; v8bf h[2]; };

__device__ __forceinline__ float gelu_f(float v) {
    return 0.5f * v * (1.0f + erff(v * 0.70710678118654752440f));
}
__device__ __forceinline__ float sigmoid_f(float v) {
    return 1.0f / (1.0f + expf(-v));
}

// A-matrix fragment (16x32 bf16) from LDS, row-major with given row stride.
// Layout (ISA 7.12.2): lane<16 -> K {c*32+0..7, c*32+16..23}; lane>=16 -> +8.
__device__ __forceinline__ v16bf load_a(const __bf16* base, int rowstride,
                                        int lane, int chunk) {
    const int row = lane & 15;
    const int k0  = chunk * 32 + ((lane & 16) ? 8 : 0);
    const __bf16* p = base + row * rowstride + k0;
    V16U u;
    u.h[0] = *(const v8bf*)(p);
    u.h[1] = *(const v8bf*)(p + 16);
    return u.v;
}

// B-matrix fragment (32x16 bf16) from pre-swizzled workspace: contiguous
// 16 bf16 (32 bytes) per lane.
__device__ __forceinline__ v16bf load_b(const __bf16* ws, int chunk, int tile,
                                        int lane) {
    return *(const v16bf*)(ws + ((((chunk << 4) + tile) << 5) + lane) * 16);
}

// Contiguous dot product: wT (global, per-thread row) . v (LDS), n multiple of 4.
__device__ __forceinline__ float dotT(const float* __restrict__ wT,
                                      const float* __restrict__ v, int n) {
    const float4* w4 = (const float4*)wT;
    const float4* v4 = (const float4*)v;
    float acc = 0.0f;
#pragma unroll 4
    for (int i = 0; i < n / 4; ++i) {
        float4 w  = w4[i];
        float4 xx = v4[i];
        acc = fmaf(w.x, xx.x, acc);
        acc = fmaf(w.y, xx.y, acc);
        acc = fmaf(w.z, xx.z, acc);
        acc = fmaf(w.w, xx.w, acc);
    }
    return acc;
}

// Block-wide sum over 256 threads via LDS tree.
__device__ __forceinline__ float block_sum(float v, float* red, int tid) {
    red[tid] = v;
    __syncthreads();
    for (int s = 128; s > 0; s >>= 1) {
        if (tid < s) red[tid] += red[tid + s];
        __syncthreads();
    }
    float r = red[0];
    __syncthreads();
    return r;
}

// Fused dual block-wide sum (one barrier tree for sum and sum-of-squares).
__device__ __forceinline__ void block_sum2(float a, float b, float* red,
                                           float* red2, int tid, float& oa,
                                           float& ob) {
    red[tid]  = a;
    red2[tid] = b;
    __syncthreads();
    for (int s = 128; s > 0; s >>= 1) {
        if (tid < s) {
            red[tid]  += red[tid + s];
            red2[tid] += red2[tid + s];
        }
        __syncthreads();
    }
    oa = red[0];
    ob = red2[0];
    __syncthreads();
}

// Pre-swizzle one (in_rows x 256) f32 weight into WMMA B-fragment-major bf16.
__global__ void swizzle_weight(const float* __restrict__ src,
                               __bf16* __restrict__ dst, int in_rows) {
    int total = in_rows * 256;
    for (int o = blockIdx.x * blockDim.x + threadIdx.x; o < total;
         o += gridDim.x * blockDim.x) {
        int e    = o & 15;
        int lane = (o >> 4) & 31;
        int tile = (o >> 9) & 15;
        int ch   = o >> 13;
        int krow = ch * 32 + ((lane & 16) ? 8 : 0) + (e < 8 ? e : e + 8);
        int col  = tile * 16 + (lane & 15);
        dst[o] = (__bf16)src[krow * 256 + col];
    }
}

// Transpose one (in_rows x 256) f32 weight -> (256 x in_rows) f32 so GEMV
// threads read contiguous rows.
__global__ void transpose_weight(const float* __restrict__ src,
                                 float* __restrict__ dst, int in_rows) {
    int total = in_rows * 256;
    for (int o = blockIdx.x * blockDim.x + threadIdx.x; o < total;
         o += gridDim.x * blockDim.x) {
        int dd = o % in_rows;
        int n  = o / in_rows;
        dst[o] = src[dd * 256 + n];
    }
}

__global__ __launch_bounds__(256) void memunit_scan(
    const float* __restrict__ x, const float* __restrict__ slot_init,
    const float* __restrict__ bq,
    const float* __restrict__ bk, const float* __restrict__ bv,
    const float* __restrict__ wg1b,
    const float* __restrict__ wg2W, const float* __restrict__ wg2b,
    const float* __restrict__ egb,
    const float* __restrict__ wcb, const float* __restrict__ wcg,
    const float* __restrict__ wcbln,
    const float* __restrict__ fub,
    const float* __restrict__ fug, const float* __restrict__ fubln,
    const float* __restrict__ ong, const float* __restrict__ onb,
    const __bf16* __restrict__ wsWk, const __bf16* __restrict__ wsWv,
    const __bf16* __restrict__ wsEg, const __bf16* __restrict__ wsWc,
    const float* __restrict__ WqT, const float* __restrict__ fuT,
    const float* __restrict__ wg1T,
    float* __restrict__ out, int S) {

    __shared__ float  Mf[K * D];                 // 16 KB recurrent state
    __shared__ __align__(32) __bf16 Mbf[K * D];  // 8 KB bf16 shadow (WMMA A)
    __shared__ __align__(16) float xs[D];
    __shared__ __align__(32) __bf16 xbf[D];
    __shared__ float  kbuf[K * D];               // k, later erase
    __shared__ float  vbuf[K * D];               // v, later write(pre-LN)
    __shared__ __align__(16) float qbuf[D];
    __shared__ __align__(16) float rbuf[D];
    __shared__ float  red[256];
    __shared__ float  red2[256];
    __shared__ float  attn[K], pers[K], evict[K], mu[K], rs[K], sml[K];

    const int tid  = threadIdx.x;
    const int lane = tid & 31;
    const int wv   = tid >> 5;
    const int b    = blockIdx.x;

    // ---- hoist per-step-invariant scalars into registers ----
    const float bq_r    = bq[tid];
    const float fub_r   = fub[tid];
    const float wg1b_r  = wg1b[tid];
    const float wg2w_r  = wg2W[tid];
    const float wg2b_r  = wg2b[0];
    const float fug_r   = fug[tid];
    const float fubln_r = fubln[tid];
    const float ong_r   = ong[tid];
    const float onb_r   = onb[tid];
    const float wcg_r   = wcg[tid];    // M-update column n == tid
    const float wcbln_r = wcbln[tid];
    const int   col0    = (wv * 2) * 16 + (lane & 15);
    const int   col1    = col0 + 16;
    const float bk0 = bk[col0], bk1 = bk[col1];
    const float bv0 = bv[col0], bv1 = bv[col1];
    const float egb0 = egb[col0], egb1 = egb[col1];
    const float wcb0 = wcb[col0], wcb1 = wcb[col1];

    const float* WqT_r  = WqT + (size_t)tid * D;       // 256 f32, contiguous
    const float* fuT_r  = fuT + (size_t)tid * 2 * D;   // 512 f32, contiguous
    const float* wg1T_r = wg1T + (size_t)tid * 2 * D;

    // ---- init state ----
    for (int rep = 0; rep < (K * D) / 256; ++rep) {
        int e   = tid + rep * 256;
        float m = slot_init[e];
        Mf[e]   = m;
        Mbf[e]  = (__bf16)m;
    }
    if (tid < K) pers[tid] = 0.0f;
    __syncthreads();

    for (int t = 0; t < S; ++t) {
        // ---- load x_t; prefetch next step's row ----
        float xv = x[((size_t)t * Bsz + b) * D + tid];
        if (t + 1 < S)
            __builtin_prefetch(&x[((size_t)(t + 1) * Bsz + b) * D + tid], 0, 3);
        xs[tid]  = xv;
        xbf[tid] = (__bf16)xv;
        __syncthreads();

        // ---- q = x_t @ Wq + bq  (GEMV, transposed weights, b128 streams) ----
        qbuf[tid] = bq_r + dotT(WqT_r, xs, D);

        // ---- k = M@Wk + bk, v = M@Wv + bv  (WMMA, 2 N-tiles per wave) ----
#pragma unroll 1
        for (int tt = 0; tt < 2; ++tt) {
            int tile = wv * 2 + tt;
            v8f ck = {};
            v8f cv = {};
#pragma unroll 2
            for (int c = 0; c < 8; ++c) {
                v16bf a  = load_a(Mbf, D, lane, c);
                v16bf fk = load_b(wsWk, c, tile, lane);
                v16bf fv = load_b(wsWv, c, tile, lane);
                ck = __builtin_amdgcn_wmma_f32_16x16x32_bf16(
                        false, a, false, fk, (short)0, ck, false, false);
                cv = __builtin_amdgcn_wmma_f32_16x16x32_bf16(
                        false, a, false, fv, (short)0, cv, false, false);
            }
            int   col   = tt ? col1 : col0;
            float biask = tt ? bk1 : bk0;
            float biasv = tt ? bv1 : bv0;
            int   rbase = (lane & 16) ? 8 : 0;
#pragma unroll
            for (int r = 0; r < 8; ++r) {
                int row = rbase + r;
                kbuf[row * D + col] = ck[r] + biask;
                vbuf[row * D + col] = cv[r] + biasv;
            }
        }
        __syncthreads();

        // ---- attn = softmax(q . k / 16) over K=16 ----
        {
            int   slot = tid >> 4, seg = tid & 15;
            float p = 0.0f;
#pragma unroll
            for (int j = 0; j < 16; ++j)
                p = fmaf(qbuf[seg * 16 + j], kbuf[slot * D + seg * 16 + j], p);
            red[tid] = p;
        }
        __syncthreads();
        if (tid < K) {
            float s = 0.0f;
            for (int j = 0; j < 16; ++j) s += red[tid * 16 + j];
            sml[tid] = s * 0.0625f;  // d^-0.5 = 1/16
        }
        __syncthreads();
        if (tid < K) {
            float m = sml[0];
            for (int j = 1; j < 16; ++j) m = fmaxf(m, sml[j]);
            red[tid] = expf(sml[tid] - m);
        }
        __syncthreads();
        if (tid < K) {
            float s = 0.0f;
            for (int j = 0; j < 16; ++j) s += red[j];
            float a   = red[tid] / s;
            attn[tid] = a;
            pers[tid] = 0.9f * pers[tid] + 0.1f * a;  // momentum update
        }
        __syncthreads();

        // ---- novelty (uniform over block) ----
        float amax = attn[0];
        for (int j = 1; j < 16; ++j) amax = fmaxf(amax, attn[j]);
        const float novelty = 1.0f - amax;

        // ---- r = attn @ v ----
        {
            float acc = 0.0f;
#pragma unroll
            for (int s2 = 0; s2 < K; ++s2)
                acc = fmaf(attn[s2], vbuf[s2 * D + tid], acc);
            rbuf[tid] = acc;
        }
        __syncthreads();

        // ---- fu branch: u = gelu(cat @ fu_W + fu_b) ----
        float u = gelu_f(fub_r + dotT(fuT_r, xs, D) + dotT(fuT_r + D, rbuf, D));

        // ---- write-gate: g_w = sigmoid(gelu(cat@wg1)@wg2 + b) * novelty ----
        float g_w;
        {
            float h = gelu_f(wg1b_r + dotT(wg1T_r, xs, D) +
                             dotT(wg1T_r + D, rbuf, D));
            float tot = block_sum(h * wg2w_r, red, tid);
            g_w = sigmoid_f(tot + wg2b_r) * novelty;
        }

        // ---- out_t = LN(u); x_aug[t] = LN(out_t + x_t) -> store ----
        {
            float s, s2;
            block_sum2(u, u * u, red, red2, tid, s, s2);
            float mean = s * (1.0f / 256.0f);
            float var  = s2 * (1.0f / 256.0f) - mean * mean;
            float o    = (u - mean) * rsqrtf(var + 1e-5f) * fug_r + fubln_r;
            float y    = o + xs[tid];
            float ys, ys2;
            block_sum2(y, y * y, red, red2, tid, ys, ys2);
            float my = ys * (1.0f / 256.0f);
            float vy = ys2 * (1.0f / 256.0f) - my * my;
            out[((size_t)t * Bsz + b) * D + tid] =
                (y - my) * rsqrtf(vy + 1e-5f) * ong_r + onb_r;
        }

        // ---- evict = softmax(-4*pers); pers += evict*g_w*0.1 ----
        if (tid < K) sml[tid] = -4.0f * pers[tid];
        __syncthreads();
        if (tid < K) {
            float m = sml[0];
            for (int j = 1; j < 16; ++j) m = fmaxf(m, sml[j]);
            red[tid] = expf(sml[tid] - m);
        }
        __syncthreads();
        if (tid < K) {
            float s = 0.0f;
            for (int j = 0; j < 16; ++j) s += red[j];
            float ev   = red[tid] / s;
            evict[tid] = ev;
            pers[tid] += ev * g_w * 0.1f;
        }
        __syncthreads();

        // ---- erase/write: sq = [x_t | M] (16x512); WMMA over 16 K-chunks ----
#pragma unroll 1
        for (int tt = 0; tt < 2; ++tt) {
            int tile = wv * 2 + tt;
            v8f ce = {};
            v8f cw = {};
#pragma unroll 2
            for (int c = 0; c < 16; ++c) {
                v16bf a  = (c < 8) ? load_a(xbf, 0, lane, c)       // broadcast x
                                   : load_a(Mbf, D, lane, c - 8);  // M rows
                v16bf fe = load_b(wsEg, c, tile, lane);
                v16bf fw = load_b(wsWc, c, tile, lane);
                ce = __builtin_amdgcn_wmma_f32_16x16x32_bf16(
                        false, a, false, fe, (short)0, ce, false, false);
                cw = __builtin_amdgcn_wmma_f32_16x16x32_bf16(
                        false, a, false, fw, (short)0, cw, false, false);
            }
            int   col   = tt ? col1 : col0;
            float biase = tt ? egb1 : egb0;
            float biasw = tt ? wcb1 : wcb0;
            int   rbase = (lane & 16) ? 8 : 0;
#pragma unroll
            for (int r = 0; r < 8; ++r) {
                int row = rbase + r;
                kbuf[row * D + col] = sigmoid_f(ce[r] + biase);  // erase
                vbuf[row * D + col] = gelu_f(cw[r] + biasw);     // write (pre-LN)
            }
        }
        __syncthreads();

        // ---- row-wise LN stats of write (per slot over d) ----
        {
            int   slot = tid >> 4, seg = tid & 15;
            float s = 0.0f, s2 = 0.0f;
#pragma unroll
            for (int j = 0; j < 16; ++j) {
                float w = vbuf[slot * D + seg * 16 + j];
                s += w;
                s2 = fmaf(w, w, s2);
            }
            red[tid]  = s;
            red2[tid] = s2;
        }
        __syncthreads();
        if (tid < K) {
            float s = 0.0f, s2 = 0.0f;
            for (int j = 0; j < 16; ++j) {
                s  += red[tid * 16 + j];
                s2 += red2[tid * 16 + j];
            }
            float mean = s * (1.0f / 256.0f);
            float var  = s2 * (1.0f / 256.0f) - mean * mean;
            mu[tid] = mean;
            rs[tid] = rsqrtf(var + 1e-5f);
        }
        __syncthreads();

        // ---- M = M*(1 - alpha*erase) + alpha*write_ln  (n == tid) ----
        for (int rep = 0; rep < (K * D) / 256; ++rep) {
            int   e    = tid + rep * 256;
            int   slot = rep;
            float w  = (vbuf[e] - mu[slot]) * rs[slot] * wcg_r + wcbln_r;
            float er = kbuf[e];
            float al = g_w * evict[slot];
            float m  = Mf[e] * (1.0f - al * er) + al * w;
            Mf[e]  = m;
            Mbf[e] = (__bf16)m;
        }
        __syncthreads();
    }

    // ---- proto = transpose(M) -> out[S*B*D + (k*B + b)*D + n] ----
    size_t base = (size_t)S * Bsz * D;
    for (int rep = 0; rep < (K * D) / 256; ++rep) {
        int e = tid + rep * 256;
        int slot = e >> 8, n = e & 255;
        out[base + ((size_t)slot * Bsz + b) * D + n] = Mf[e];
    }
}

extern "C" void kernel_launch(void* const* d_in, const int* in_sizes, int n_in,
                              void* d_out, int out_size, void* d_ws,
                              size_t ws_size, hipStream_t stream) {
    const float* x     = (const float*)d_in[0];
    const float* slot  = (const float*)d_in[1];
    const float* Wq    = (const float*)d_in[2];
    const float* bq    = (const float*)d_in[3];
    const float* Wk    = (const float*)d_in[4];
    const float* bk    = (const float*)d_in[5];
    const float* Wv    = (const float*)d_in[6];
    const float* bv    = (const float*)d_in[7];
    const float* wg1W  = (const float*)d_in[8];
    const float* wg1b  = (const float*)d_in[9];
    const float* wg2W  = (const float*)d_in[10];
    const float* wg2b  = (const float*)d_in[11];
    const float* egW   = (const float*)d_in[12];
    const float* egb   = (const float*)d_in[13];
    const float* wcW   = (const float*)d_in[14];
    const float* wcb   = (const float*)d_in[15];
    const float* wcg   = (const float*)d_in[16];
    const float* wcbln = (const float*)d_in[17];
    const float* fuW   = (const float*)d_in[18];
    const float* fub   = (const float*)d_in[19];
    const float* fug   = (const float*)d_in[20];
    const float* fubln = (const float*)d_in[21];
    const float* ong   = (const float*)d_in[22];
    const float* onb   = (const float*)d_in[23];

    const int S = in_sizes[0] / (Bsz * D);  // 2048

    // Workspace: bf16 B-fragment-swizzled WMMA weights, then f32 transposed
    // GEMV weights.
    __bf16* wsWk = (__bf16*)d_ws;
    __bf16* wsWv = wsWk + D * D;
    __bf16* wsEg = wsWv + D * D;
    __bf16* wsWc = wsEg + 2 * D * D;
    float*  WqT  = (float*)(wsWc + 2 * D * D);   // 786432 B in, 16B aligned
    float*  fuT  = WqT + D * D;
    float*  wg1T = fuT + 2 * D * D;

    swizzle_weight<<<64, 256, 0, stream>>>(Wk, wsWk, D);
    swizzle_weight<<<64, 256, 0, stream>>>(Wv, wsWv, D);
    swizzle_weight<<<128, 256, 0, stream>>>(egW, wsEg, 2 * D);
    swizzle_weight<<<128, 256, 0, stream>>>(wcW, wsWc, 2 * D);
    transpose_weight<<<64, 256, 0, stream>>>(Wq, WqT, D);
    transpose_weight<<<128, 256, 0, stream>>>(fuW, fuT, 2 * D);
    transpose_weight<<<128, 256, 0, stream>>>(wg1W, wg1T, 2 * D);

    memunit_scan<<<Bsz, 256, 0, stream>>>(
        x, slot, bq, bk, bv, wg1b, wg2W, wg2b, egb, wcb, wcg, wcbln, fub, fug,
        fubln, ong, onb, wsWk, wsWv, wsEg, wsWc, WqT, fuT, wg1T, (float*)d_out,
        S);
}
